// Transformer_performer_30039001268325
// MI455X (gfx1250) — matrix operand order repaired and checked by reading, not compiled, not run on hardware
//
#include <hip/hip_runtime.h>
#include <hip/hip_bf16.h>

typedef __attribute__((ext_vector_type(16))) __bf16 v16bf;
typedef __attribute__((ext_vector_type(8)))  float  v8f;

union FragU { unsigned int u[8]; v16bf v; };

__device__ inline unsigned short f2bf(float f) {
    unsigned int u = __float_as_uint(f);
    u += 0x7FFFu + ((u >> 16) & 1u);           // round-to-nearest-even
    return (unsigned short)(u >> 16);
}
__device__ inline float bf2f(unsigned short h) {
    return __uint_as_float(((unsigned int)h) << 16);
}

// ---------------- WMMA GEMM ----------------
// C[M,N] = A[M,K] @ B[K,N] (+epilogue). LDS: As[m][k], Bs[n][k] (B transposed
// on store) so fragment gathers are contiguous ds_load_b128 along K.
// Register-staged double buffering: global->regs for tile t+1 issued before the
// WMMAs of tile t; regs->LDS(alt buffer) after; one barrier per iteration.
#define BM 128
#define BN 128
#define BKT 64
#define LDK 72   // padded K stride (elements)

// 16-bit A fragment 16x32: lanes 0-15 K in {0..7,16..23}; lanes 16-31 K in {8..15,24..31}
__device__ inline v16bf frag_a(const unsigned short* As, int m, int half, int kk) {
    FragU f;
    const unsigned int* rowp = (const unsigned int*)(As + m * LDK);
#pragma unroll
    for (int j = 0; j < 8; ++j) {
        int k2 = ((j >> 2) << 4) + (half << 3) + ((j & 3) << 1) + kk;
        f.u[j] = rowp[k2 >> 1];
    }
    return f.v;
}
// 16-bit B fragment 32x16: lanes 0-15 K in {0..7,16..23}; lanes 16-31 K in {8..15,24..31}
__device__ inline v16bf frag_b(const unsigned short* Bs, int n, int half, int kk) {
    FragU f;
    const unsigned int* rowp = (const unsigned int*)(Bs + n * LDK);
#pragma unroll
    for (int j = 0; j < 8; ++j) {
        int k = ((j >> 2) << 4) + (half << 3) + ((j & 3) << 1) + kk;
        f.u[j] = rowp[k >> 1];
    }
    return f.v;
}

// EPI: 0 = (+bias if non-null), 1 = phi: exp(acc - 0.5*aux[row])/16, 2 = acc/aux[row] + resid
template <bool AF32, bool BF32, bool TRANSA, bool TRANSB, int EPI>
__global__ __launch_bounds__(256) void gemm_wmma_kernel(
    const void* __restrict__ Ag, const void* __restrict__ Bg,
    unsigned short* __restrict__ C,
    const float* __restrict__ bias,
    const float* __restrict__ rowAux,
    const unsigned short* __restrict__ resid,
    int K, int lda, int ldb, int ldc,
    long aBatch, long bBatch, long cBatch, long auxBatch, long residBatch)
{
    __shared__ unsigned short As[2][BM * LDK];
    __shared__ unsigned short Bs[2][BN * LDK];

    const int tid   = threadIdx.x;
    const int lane  = tid & 31;
    const int wave  = tid >> 5;
    const int waveM = wave & 3;   // 4 waves along M -> 32 rows each
    const int waveN = wave >> 2;  // 2 waves along N -> 64 cols each
    const int half  = lane >> 4;
    const int lm    = lane & 15;

    const int b       = blockIdx.z;
    const int rowBase = blockIdx.x * BM;
    const int colBase = blockIdx.y * BN;
    const long aOff = (long)b * aBatch;
    const long bOff = (long)b * bBatch;

    const float*          Af = (const float*)Ag;
    const unsigned short* Ab = (const unsigned short*)Ag;
    const float*          Bf = (const float*)Bg;
    const unsigned short* Bb = (const unsigned short*)Bg;

    unsigned int regA[16], regB[16];   // staged bf16x2 for one 128x64 tile each

    // ---- global -> registers (bf16-packed) ----
    auto loadA = [&](int kb) {
        long base = TRANSA ? (aOff + (long)kb * lda + rowBase)   // global [K][M], width 128
                           : (aOff + (long)rowBase * lda + kb);  // global [M][K], width 64
#pragma unroll
        for (int i = 0; i < 8; ++i) {
            int e = (tid + (i << 8)) << 2;
            int r = TRANSA ? (e >> 7) : (e >> 6);
            int c = TRANSA ? (e & 127) : (e & 63);
            long g = base + (long)r * lda + c;
            if (AF32) {
                const float4 t = *(const float4*)(Af + g);
                regA[2 * i]     = (unsigned)f2bf(t.x) | ((unsigned)f2bf(t.y) << 16);
                regA[2 * i + 1] = (unsigned)f2bf(t.z) | ((unsigned)f2bf(t.w) << 16);
            } else {
                const uint2 t = *(const uint2*)(Ab + g);
                regA[2 * i] = t.x; regA[2 * i + 1] = t.y;
            }
        }
    };
    auto loadB = [&](int kb) {
        long base = TRANSB ? (bOff + (long)colBase * ldb + kb)   // global [N][K], width 64
                           : (bOff + (long)kb * ldb + colBase);  // global [K][N], width 128
#pragma unroll
        for (int i = 0; i < 8; ++i) {
            int e = (tid + (i << 8)) << 2;
            int r = TRANSB ? (e >> 6) : (e >> 7);
            int c = TRANSB ? (e & 63) : (e & 127);
            long g = base + (long)r * ldb + c;
            if (BF32) {
                const float4 t = *(const float4*)(Bf + g);
                regB[2 * i]     = (unsigned)f2bf(t.x) | ((unsigned)f2bf(t.y) << 16);
                regB[2 * i + 1] = (unsigned)f2bf(t.z) | ((unsigned)f2bf(t.w) << 16);
            } else {
                const uint2 t = *(const uint2*)(Bb + g);
                regB[2 * i] = t.x; regB[2 * i + 1] = t.y;
            }
        }
    };
    // ---- registers -> LDS ----
    auto storeA = [&](int buf) {
#pragma unroll
        for (int i = 0; i < 8; ++i) {
            int e = (tid + (i << 8)) << 2;
            if (TRANSA) {              // transpose-store into As[m][k]
                int r = e >> 7, c = e & 127;
#pragma unroll
                for (int j = 0; j < 4; ++j)
                    As[buf][(c + j) * LDK + r] =
                        (unsigned short)(regA[2 * i + (j >> 1)] >> ((j & 1) * 16));
            } else {                   // direct, 8-byte packed
                int r = e >> 6, c = e & 63;
                uint2 t; t.x = regA[2 * i]; t.y = regA[2 * i + 1];
                *(uint2*)(&As[buf][r * LDK + c]) = t;
            }
        }
    };
    auto storeB = [&](int buf) {
#pragma unroll
        for (int i = 0; i < 8; ++i) {
            int e = (tid + (i << 8)) << 2;
            if (TRANSB) {              // direct, 8-byte packed into Bs[n][k]
                int r = e >> 6, c = e & 63;
                uint2 t; t.x = regB[2 * i]; t.y = regB[2 * i + 1];
                *(uint2*)(&Bs[buf][r * LDK + c]) = t;
            } else {                   // transpose-store into Bs[n][k]
                int r = e >> 7, c = e & 127;
#pragma unroll
                for (int j = 0; j < 4; ++j)
                    Bs[buf][(c + j) * LDK + r] =
                        (unsigned short)(regB[2 * i + (j >> 1)] >> ((j & 1) * 16));
            }
        }
    };
    // ---- L2 warm-up two tiles ahead (global_prefetch_b8) ----
    auto prefetch = [&](int kb) {
        {
            long base = TRANSA ? (aOff + (long)kb * lda + rowBase)
                               : (aOff + (long)rowBase * lda + kb);
            int e = tid << 2;
            int r = TRANSA ? (e >> 7) : (e >> 6);
            int c = TRANSA ? (e & 127) : (e & 63);
            long g = base + (long)r * lda + c;
            __builtin_prefetch(AF32 ? (const void*)(Af + g) : (const void*)(Ab + g), 0, 1);
        }
        {
            long base = TRANSB ? (bOff + (long)colBase * ldb + kb)
                               : (bOff + (long)kb * ldb + colBase);
            int e = tid << 2;
            int r = TRANSB ? (e >> 6) : (e >> 7);
            int c = TRANSB ? (e & 63) : (e & 127);
            long g = base + (long)r * ldb + c;
            __builtin_prefetch(BF32 ? (const void*)(Bf + g) : (const void*)(Bb + g), 0, 1);
        }
    };

    v8f acc[2][4];
#pragma unroll
    for (int tm = 0; tm < 2; ++tm)
#pragma unroll
        for (int tn = 0; tn < 4; ++tn)
#pragma unroll
            for (int r = 0; r < 8; ++r) acc[tm][tn][r] = 0.0f;

    auto compute = [&](int buf) {
#pragma unroll
        for (int kk = 0; kk < BKT; kk += 32) {
            v16bf af[2], bfr[4];
#pragma unroll
            for (int tm = 0; tm < 2; ++tm)
                af[tm] = frag_a(As[buf], waveM * 32 + tm * 16 + lm, half, kk);
#pragma unroll
            for (int tn = 0; tn < 4; ++tn)
                bfr[tn] = frag_b(Bs[buf], waveN * 64 + tn * 16 + lm, half, kk);
#pragma unroll
            for (int tm = 0; tm < 2; ++tm)
#pragma unroll
                for (int tn = 0; tn < 4; ++tn)
                    acc[tm][tn] = __builtin_amdgcn_wmma_f32_16x16x32_bf16(
                        false, af[tm], false, bfr[tn], (short)0, acc[tm][tn], false, false);
        }
    };

    // ---- pipelined main loop ----
    const int T = K / BKT;
    loadA(0); loadB(0);
    storeA(0); storeB(0);
    int cur = 0;
    for (int t = 0; t < T; ++t) {
        __syncthreads();
        if (t + 1 < T) {
            loadA((t + 1) * BKT);
            loadB((t + 1) * BKT);
            if (t + 2 < T) prefetch((t + 2) * BKT);
        }
        compute(cur);
        if (t + 1 < T) {
            storeA(cur ^ 1);
            storeB(cur ^ 1);
            cur ^= 1;
        }
    }

    // ---- epilogue + store bf16 (aux/bias hoisted, branch-free inner loop) ----
    const long cOff = (long)b * cBatch;
#pragma unroll
    for (int tm = 0; tm < 2; ++tm) {
        const int row0 = rowBase + waveM * 32 + tm * 16 + half * 8;
        float aux[8];
        if (EPI != 0) {
#pragma unroll
            for (int r = 0; r < 8; ++r)
                aux[r] = rowAux[(long)b * auxBatch + row0 + r];
        }
#pragma unroll
        for (int tn = 0; tn < 4; ++tn) {
            const int col = colBase + waveN * 64 + tn * 16 + lm;
            float bv = 0.0f;
            if (EPI == 0 && bias) bv = bias[col];
#pragma unroll
            for (int r = 0; r < 8; ++r) {
                const int rr = row0 + r;
                float v = acc[tm][tn][r];
                if (EPI == 0) {
                    v += bv;
                } else if (EPI == 1) {
                    v = __expf(v - 0.5f * aux[r]) * 0.0625f;
                } else {
                    v = v / aux[r]
                        + bf2f(resid[(long)b * residBatch + (long)rr * ldc + col]);
                }
                C[cOff + (long)rr * ldc + col] = f2bf(v);
            }
        }
    }
}

// ---------------- small helper kernels ----------------
__global__ __launch_bounds__(256) void rowsumsq_kernel(
    const unsigned short* __restrict__ x, float* __restrict__ out, int ncols)
{
    __shared__ float red[256];
    long row = blockIdx.x;
    float v = 0.f;
    for (int c = threadIdx.x; c < ncols; c += 256) {
        float a = bf2f(x[row * ncols + c]);
        v += a * a;
    }
    red[threadIdx.x] = v; __syncthreads();
    for (int s = 128; s > 0; s >>= 1) {
        if (threadIdx.x < (unsigned)s) red[threadIdx.x] += red[threadIdx.x + s];
        __syncthreads();
    }
    if (threadIdx.x == 0) out[row] = red[0];
}

__global__ __launch_bounds__(256) void colsum_kernel(
    const unsigned short* __restrict__ kp, float* __restrict__ ksum, int L, int M)
{
    int b = blockIdx.x, m = threadIdx.x;
    const unsigned short* p = kp + (long)b * L * M + m;
    float s = 0.f;
    for (int l = 0; l < L; ++l) s += bf2f(p[(long)l * M]);
    ksum[b * M + m] = s;
}

__global__ __launch_bounds__(256) void denom_kernel(
    const unsigned short* __restrict__ qp, const float* __restrict__ ksum,
    float* __restrict__ denom, int M, int L)
{
    __shared__ float red[256];
    long row = blockIdx.x;
    int  b   = (int)(row / L);
    red[threadIdx.x] = bf2f(qp[row * M + threadIdx.x]) * ksum[b * M + threadIdx.x];
    __syncthreads();
    for (int s = 128; s > 0; s >>= 1) {
        if (threadIdx.x < (unsigned)s) red[threadIdx.x] += red[threadIdx.x + s];
        __syncthreads();
    }
    if (threadIdx.x == 0) denom[row] = red[0];
}

__global__ __launch_bounds__(256) void final_kernel(
    const unsigned short* __restrict__ outbf, const float* __restrict__ Wfin,
    const float* __restrict__ bfin, float* __restrict__ y, int Ktot)
{
    __shared__ float red[256];
    int b = blockIdx.x, tid = threadIdx.x;
    float acc[10];
#pragma unroll
    for (int c = 0; c < 10; ++c) acc[c] = 0.f;
    const unsigned short* xr = outbf + (long)b * Ktot;
    for (int k = tid; k < Ktot; k += 256) {
        float a = bf2f(xr[k]);
        const float* w = Wfin + (long)k * 10;
#pragma unroll
        for (int c = 0; c < 10; ++c) acc[c] += a * w[c];
    }
    for (int c = 0; c < 10; ++c) {
        red[tid] = acc[c]; __syncthreads();
        for (int s = 128; s > 0; s >>= 1) {
            if (tid < s) red[tid] += red[tid + s];
            __syncthreads();
        }
        if (tid == 0) y[b * 10 + c] = red[0] + bfin[c];
        __syncthreads();
    }
}

// ---------------- launch ----------------
extern "C" void kernel_launch(void* const* d_in, const int* in_sizes, int n_in,
                              void* d_out, int out_size, void* d_ws, size_t ws_size,
                              hipStream_t stream)
{
    const float* x     = (const float*)d_in[0];
    const float* W_lin = (const float*)d_in[1];
    const float* b_lin = (const float*)d_in[2];
    const float* Wq    = (const float*)d_in[3];
    const float* Wk    = (const float*)d_in[4];
    const float* Wv    = (const float*)d_in[5];
    const float* W_add = (const float*)d_in[6];
    const float* b_add = (const float*)d_in[7];
    const float* W_fin = (const float*)d_in[8];
    const float* b_fin = (const float*)d_in[9];
    const float* rf    = (const float*)d_in[10];
    float* y = (float*)d_out;

    const int  B = 64, L = 1024, DMAX = 512, DLEN = 256, DM = 256, M = 256;
    const long ROWS = (long)B * L;  // 65536

    char* ws = (char*)d_ws;
    size_t off = 0;
    auto alloc = [&](size_t bytes) -> void* {
        void* p = ws + off; off += (bytes + 255) & ~(size_t)255; return p;
    };
    unsigned short* e_bf  = (unsigned short*)alloc(ROWS * DLEN * 2);
    unsigned short* q_bf  = (unsigned short*)alloc(ROWS * DM * 2);
    unsigned short* k_bf  = (unsigned short*)alloc(ROWS * DM * 2);
    unsigned short* v_bf  = (unsigned short*)alloc(ROWS * DM * 2);
    unsigned short* resid = (unsigned short*)alloc(ROWS * DM * 2);
    unsigned short* kv_bf = (unsigned short*)alloc((long)B * M * DM * 2);
    float* qnorm = (float*)alloc(ROWS * 4);
    float* knorm = (float*)alloc(ROWS * 4);
    float* ksum  = (float*)alloc((long)B * M * 4);
    float* denom = (float*)alloc(ROWS * 4);
    // lifetime-based aliases (each fully rewritten before read)
    unsigned short* qp_bf  = e_bf;  // e dead after q/k/v/resid GEMMs
    unsigned short* kp_bf  = q_bf;  // q dead after qp GEMM
    unsigned short* out_bf = k_bf;  // k dead after kp GEMM

    dim3 blk(256);
    dim3 gridRows((int)(ROWS / 128), DM / 128, 1);

    // e = x @ W_lin + b_lin   [65536,512]x[512,256]
    gemm_wmma_kernel<true, true, false, false, 0><<<dim3((int)(ROWS / 128), DLEN / 128, 1), blk, 0, stream>>>(
        x, W_lin, e_bf, b_lin, nullptr, nullptr, DMAX, DMAX, DLEN, DLEN, 0, 0, 0, 0, 0);

    // q, k, v, resid = e @ {Wq,Wk,Wv,W_add(+b_add)}
    gemm_wmma_kernel<false, true, false, false, 0><<<gridRows, blk, 0, stream>>>(
        e_bf, Wq, q_bf, nullptr, nullptr, nullptr, DLEN, DLEN, DM, DM, 0, 0, 0, 0, 0);
    gemm_wmma_kernel<false, true, false, false, 0><<<gridRows, blk, 0, stream>>>(
        e_bf, Wk, k_bf, nullptr, nullptr, nullptr, DLEN, DLEN, DM, DM, 0, 0, 0, 0, 0);
    gemm_wmma_kernel<false, true, false, false, 0><<<gridRows, blk, 0, stream>>>(
        e_bf, Wv, v_bf, nullptr, nullptr, nullptr, DLEN, DLEN, DM, DM, 0, 0, 0, 0, 0);
    gemm_wmma_kernel<false, true, false, false, 0><<<gridRows, blk, 0, stream>>>(
        e_bf, W_add, resid, b_add, nullptr, nullptr, DLEN, DLEN, DM, DM, 0, 0, 0, 0, 0);

    // row ||q||^2, ||k||^2
    rowsumsq_kernel<<<(int)ROWS, blk, 0, stream>>>(q_bf, qnorm, DM);
    rowsumsq_kernel<<<(int)ROWS, blk, 0, stream>>>(k_bf, knorm, DM);

    // qp = exp(q @ rf^T - 0.5*||q||^2)/16 ; kp likewise (rf global is [M][D] => TRANSB)
    gemm_wmma_kernel<false, true, false, true, 1><<<gridRows, blk, 0, stream>>>(
        q_bf, rf, qp_bf, nullptr, qnorm, nullptr, DM, DM, DM, M, 0, 0, 0, 0, 0);
    gemm_wmma_kernel<false, true, false, true, 1><<<gridRows, blk, 0, stream>>>(
        k_bf, rf, kp_bf, nullptr, knorm, nullptr, DM, DM, DM, M, 0, 0, 0, 0, 0);

    // ksum[b][m] = sum_l kp[b][l][m]
    colsum_kernel<<<B, blk, 0, stream>>>(kp_bf, ksum, L, M);

    // kv[b] = kp[b]^T @ v[b]   (A = kp stored [L][M] => TRANSA)
    gemm_wmma_kernel<false, false, true, false, 0><<<dim3(M / 128, DM / 128, B), blk, 0, stream>>>(
        kp_bf, v_bf, kv_bf, nullptr, nullptr, nullptr, L, M, DM, DM,
        (long)L * M, (long)L * DM, (long)M * DM, 0, 0);

    // denom[b][l] = qp[b][l] . ksum[b]
    denom_kernel<<<(int)ROWS, blk, 0, stream>>>(qp_bf, ksum, denom, M, L);

    // out[b] = qp[b] @ kv[b] / denom + resid
    gemm_wmma_kernel<false, false, false, false, 2><<<dim3(L / 128, DM / 128, B), blk, 0, stream>>>(
        qp_bf, kv_bf, out_bf, nullptr, denom, resid, M, M, DM, DM,
        (long)L * M, (long)M * DM, (long)L * DM, (long)L, (long)L * DM);

    // y = out_flat @ W_fin + b_fin
    final_kernel<<<B, blk, 0, stream>>>(out_bf, W_fin, b_fin, y, L * DM);
}
